// ExtractModel_40862318854753
// MI455X (gfx1250) — compile-verified, matrix operand order checked by reading, not run on hardware
//
#include <hip/hip_runtime.h>
#include <hip/hip_bf16.h>

#define MSL 10
#define MTL 10
#define BIGV 99.9f
#define MATCH_THRESH 0.05f

typedef __bf16 v16bf __attribute__((ext_vector_type(16)));
typedef float  v8f   __attribute__((ext_vector_type(8)));

union ABFrag { v16bf v; uint4 u[2]; };

__device__ __forceinline__ unsigned short f2bf(float f) {
  unsigned int u = __float_as_uint(f);
  u += 0x7FFFu + ((u >> 16) & 1u);          // round-to-nearest-even
  return (unsigned short)(u >> 16);
}

// Load one 16x32-K fragment set (8 K-steps) for a 16x256 bf16 tile at bb.
__device__ __forceinline__ void loadB(ABFrag (&dst)[8],
                                      const unsigned short* __restrict__ bb,
                                      int kg) {
#pragma unroll
  for (int t = 0; t < 8; ++t) {
    dst[t].u[0] = *(const uint4*)(bb + t * 32 + kg);
    dst[t].u[1] = *(const uint4*)(bb + t * 32 + 16 + kg);
  }
}

// 8-step K-accumulation WMMA chain: C = A(16x256) * B(256x16)
__device__ __forceinline__ v8f mmchain(const ABFrag (&a)[8], const ABFrag (&b)[8]) {
  v8f c = {};
#pragma unroll
  for (int t = 0; t < 8; ++t) {
    c = __builtin_amdgcn_wmma_f32_16x16x32_bf16(false, a[t].v, false, b[t].v,
                                                (short)0, c, false, false);
  }
  return c;
}

// Operands are pre-normalized, so c IS the cosine: diff = (1-cos)/2.
// Pure VALU + ds_store epilogue (no global loads -> no loadcnt drain).
__device__ __forceinline__ void epilogue(const v8f& c, int vv, int j, int rbase,
                                         float* __restrict__ myd) {
  if (j < MTL) {
#pragma unroll
    for (int r = 0; r < 8; ++r) {
      int i = rbase + r;
      if (i < MSL) myd[vv * 100 + j * 10 + i] = 0.5f - 0.5f * c[r];
    }
  }
}

// ---------------------------------------------------------------------------
// Row inverse L2-norm: dst[vec] = 1 / (||src[vec*256 .. +255]|| + 1e-8)
// ---------------------------------------------------------------------------
__global__ __launch_bounds__(256) void rowinvnorm_k(const float* __restrict__ src,
                                                    float* __restrict__ dst) {
  __shared__ float red[256];
  const int vec = blockIdx.x;
  const int tid = threadIdx.x;
  float x = src[(size_t)vec * 256 + tid];
  red[tid] = x * x;
  __syncthreads();
  for (int off = 128; off > 0; off >>= 1) {
    if (tid < off) red[tid] += red[tid + off];
    __syncthreads();
  }
  if (tid == 0) dst[vec] = 1.0f / (sqrtf(red[0]) + 1e-8f);
}

// ---------------------------------------------------------------------------
// Pack ext (gathered word_repr), pre-scaled by 1/nx, into bf16;
// padded 10 -> 16 rows per (b,s). Rows i>=10 are zero.
// ---------------------------------------------------------------------------
__global__ __launch_bounds__(256) void pack_ext_k(const float* __restrict__ word,
                                                  const float* __restrict__ rnormw,
                                                  unsigned short* __restrict__ extbf) {
  int idx  = blockIdx.x * 256 + threadIdx.x;   // 192*16*256 = 786432 exact
  int k    = idx & 255;
  int rowg = idx >> 8;
  int i    = rowg & 15;
  int bs   = rowg >> 4;                        // 0..191
  int b    = bs / 48, s = bs - b * 48;
  float val = 0.0f;
  if (i < MSL) {
    int p = s + i; if (p > 47) p = 47;
    val = word[((size_t)(b * 48 + p)) * 256 + k] * rnormw[b * 48 + p];
  }
  extbf[idx] = f2bf(val);
}

// ---------------------------------------------------------------------------
// Pack vocab_repr, pre-scaled by 1/ny, into bf16; padded 10 -> 16 rows per v.
// ---------------------------------------------------------------------------
__global__ __launch_bounds__(256) void pack_voc_k(const float* __restrict__ voc,
                                                  const float* __restrict__ rnormv,
                                                  unsigned short* __restrict__ vocbf) {
  int idx  = blockIdx.x * 256 + threadIdx.x;   // 1000*16*256 = 4096000 exact
  int k    = idx & 255;
  int rowg = idx >> 8;
  int j    = rowg & 15;
  int v    = rowg >> 4;
  float val = 0.0f;
  if (j < MTL) {
    int rj = v * MTL + j;
    val = voc[(size_t)rj * 256 + k] * rnormv[rj];
  }
  vocbf[idx] = f2bf(val);
}

// ---------------------------------------------------------------------------
// Main fused kernel: one workgroup (8 waves) per (b,s).
// Double-buffered B tiles: the WMMA chain on buffer k overlaps the global
// loads filling buffer k^1 (in-order load returns -> partial s_wait_loadcnt).
// Operands are pre-normalized, so the epilogue is loads-free. The banded
// edit-distance DP runs per vocab entry on LDS-staged diff tiles.
// ---------------------------------------------------------------------------
__global__ __launch_bounds__(256) void dp_main_k(
    const unsigned short* __restrict__ extbf,   // [192][16][256] (normalized)
    const unsigned short* __restrict__ vocbf,   // [1000][16][256] (normalized)
    const int*  __restrict__ vocab_length,      // [1000]
    float* __restrict__ valmin,                 // [192][10]
    int*   __restrict__ argv)                   // [192][10]
{
  __shared__ float lds[8 * 16 * 100];           // 51200 B: per-wave diff tiles
  const int tid  = threadIdx.x;
  const int wave = tid >> 5;
  const int lane = tid & 31;
  const int bsid = blockIdx.x;                  // b*48 + s

  const int row = lane & 15;                    // M row (A) / N col (B)
  const int kg  = (lane >> 4) << 3;             // K sub-group: 0 or 8
  const int rbase = kg;                         // C rows this lane holds

  // Resident A fragments for this (b,s): 8 K-steps x v16bf (64 VGPRs)
  ABFrag a[8];
  loadB(a, extbf + ((size_t)(bsid * 16 + row)) * 256, kg);

  float best[10];
  int   bestv[10];
#pragma unroll
  for (int i = 0; i < 10; ++i) { best[i] = 1e30f; bestv[i] = 0; }

  float* myd = lds + wave * 1600;               // 16 tiles x 100 floats
  const int j = lane & 15;

  // Software pipeline: b0 holds the current tile, b1 the in-flight tile.
  ABFrag b0[8], b1[8];
  {
    int v0 = wave * 16; if (v0 > 999) v0 = 999;
    loadB(b0, vocbf + ((size_t)(v0 * 16 + row)) * 256, kg);
  }

  for (int cc = 0; cc < 8; ++cc) {
    const int vbase = (wave + 8 * cc) * 16;     // chunks 0..63 cover 1024 >= 1000

    // Hoisted: this chunk's DP inputs. ~128 B-tile loads issue after these,
    // so their waits are free by the time the DP runs.
    const int vdp = vbase + lane;
    const int lvq = vocab_length[vdp > 999 ? 999 : vdp];

    for (int vv = 0; vv < 16; vv += 2) {
      int vb = vbase + vv + 1; if (vb > 999) vb = 999;

      loadB(b1, vocbf + ((size_t)(vb * 16 + row)) * 256, kg);   // fill next
      v8f c0 = mmchain(a, b0);                                  // compute cur
      epilogue(c0, vv, j, rbase, myd);

      int vn = (vv == 14) ? (vbase + 128) : (vbase + vv + 2);   // next chunk head
      if (vn > 999) vn = 999;
      loadB(b0, vocbf + ((size_t)(vn * 16 + row)) * 256, kg);   // fill next
      v8f c1 = mmchain(a, b1);                                  // compute cur
      epilogue(c1, vv + 1, j, rbase, myd);
    }
    __syncthreads();

    if (lane < 16 && vdp < 1000) {
      const float* dm = myd + lane * 100;       // dm[(j-1)*10 + (i-1)]
      float fprev[MTL + 1], fcur[MTL + 1];
#pragma unroll
      for (int jj = 0; jj <= MTL; ++jj) fprev[jj] = (float)jj;
#pragma unroll
      for (int i = 1; i <= MSL; ++i) {
        fcur[0] = (float)i;
#pragma unroll
        for (int jj = 1; jj <= MTL; ++jj) {
          const int jlo = (i - 2 > 1) ? (i - 2) : 1;
          const int jhi = (i + 1 < MTL) ? (i + 1) : MTL;
          if (jj >= jlo && jj <= jhi) {
            float dij = dm[(jj - 1) * 10 + (i - 1)];
            float m = fprev[jj] + 1.0f;
            m = fminf(m, fcur[jj - 1] + 1.0f);
            m = fminf(m, fprev[jj - 1] + dij);
            fcur[jj] = m;
          } else {
            fcur[jj] = BIGV;
          }
        }
        float ft = BIGV;
#pragma unroll
        for (int jj = 1; jj <= MTL; ++jj) if (lvq == jj) ft = fcur[jj];
        if (ft < best[i - 1]) { best[i - 1] = ft; bestv[i - 1] = vdp; }
#pragma unroll
        for (int jj = 0; jj <= MTL; ++jj) fprev[jj] = fcur[jj];
      }
    }
    __syncthreads();
  }

  // Cross-thread min/argmin reduction (reuse LDS: 2560 floats + 2560 ints)
  float* bvals = lds;
  int*   bidx  = (int*)(lds + 2560);
#pragma unroll
  for (int i = 0; i < 10; ++i) {
    bvals[tid * 10 + i] = best[i];
    bidx [tid * 10 + i] = bestv[i];
  }
  __syncthreads();
  if (tid < 10) {
    float m = 1e30f; int mv = 0;
    for (int t = 0; t < 256; ++t) {
      float val = bvals[t * 10 + tid];
      int   vid = bidx [t * 10 + tid];
      if (val < m || (val == m && vid < mv)) { m = val; mv = vid; }
    }
    valmin[bsid * 10 + tid] = m;
    argv  [bsid * 10 + tid] = mv;
  }
}

// ---------------------------------------------------------------------------
// Finalize: per batch, apply viability mask, score, argmax (first-index ties),
// write 16 floats: scores[4], starts[4], ends[4], matched_any[4].
// ---------------------------------------------------------------------------
__global__ __launch_bounds__(256) void finalize_k(
    const float* __restrict__ valmin, const int* __restrict__ argv,
    const int* __restrict__ vlen, const int* __restrict__ lengths,
    float* __restrict__ out)
{
  __shared__ float sv[256];
  __shared__ int   si[256];
  __shared__ int   sm[256];
  const int b = blockIdx.x, tid = threadIdx.x;
  const int len = lengths[b];
  float msc = -1.0f; int midx = 0; int many = 0;
  for (int e = tid; e < 48 * MSL; e += 256) {
    int s = e / MSL, i = e - s * MSL;
    float bv = valmin[(b * 48 + s) * 10 + i];
    int   v  = argv  [(b * 48 + s) * 10 + i];
    bool viable  = (s + i) < len;
    bool matched = viable && (bv < MATCH_THRESH);
    float score  = matched ? (float)vlen[v] * (1.0f - bv) : 0.0f;
    many |= matched ? 1 : 0;
    if (score > msc || (score == msc && e < midx)) { msc = score; midx = e; }
  }
  sv[tid] = msc; si[tid] = midx; sm[tid] = many;
  __syncthreads();
  for (int off = 128; off > 0; off >>= 1) {
    if (tid < off) {
      float ov = sv[tid + off]; int oi = si[tid + off];
      if (ov > sv[tid] || (ov == sv[tid] && oi < si[tid])) { sv[tid] = ov; si[tid] = oi; }
      sm[tid] |= sm[tid + off];
    }
    __syncthreads();
  }
  if (tid == 0) {
    int idx = si[0];
    out[b]      = sv[0];
    out[4 + b]  = (float)(idx / MSL);
    out[8 + b]  = (float)(idx % MSL + idx / MSL);
    out[12 + b] = sm[0] ? 1.0f : 0.0f;
  }
}

// ---------------------------------------------------------------------------
extern "C" void kernel_launch(void* const* d_in, const int* in_sizes, int n_in,
                              void* d_out, int out_size, void* d_ws, size_t ws_size,
                              hipStream_t stream) {
  const float* word    = (const float*)d_in[0];  // (4,48,256) f32
  const float* vocab   = (const float*)d_in[1];  // (1000,10,256) f32
  const int*   lengths = (const int*)d_in[2];    // (4,) i32
  const int*   vlen    = (const int*)d_in[3];    // (1000,) i32

  // Workspace layout (~9.9 MB total)
  unsigned short* extbf = (unsigned short*)d_ws;                 // 192*16*256
  unsigned short* vocbf = extbf + (size_t)192 * 16 * 256;        // 1000*16*256
  float* rnormw = (float*)(vocbf + (size_t)1000 * 16 * 256);     // 192
  float* rnormv = rnormw + 192;                                  // 10000
  float* valmin = rnormv + 10000;                                // 1920
  int*   argvp  = (int*)(valmin + 1920);                         // 1920

  rowinvnorm_k<<<192,   256, 0, stream>>>(word,  rnormw);
  rowinvnorm_k<<<10000, 256, 0, stream>>>(vocab, rnormv);
  pack_ext_k<<<3072,  256, 0, stream>>>(word,  rnormw, extbf);
  pack_voc_k<<<16000, 256, 0, stream>>>(vocab, rnormv, vocbf);
  dp_main_k<<<192, 256, 0, stream>>>(extbf, vocbf, vlen, valmin, argvp);
  finalize_k<<<4, 256, 0, stream>>>(valmin, argvp, vlen, lengths, (float*)d_out);
}
